// GCNLayer_1151051235750
// MI455X (gfx1250) — matrix operand order repaired and checked by reading, not compiled, not run on hardware
//
#include <hip/hip_runtime.h>

typedef __attribute__((ext_vector_type(16))) __bf16 v16bf;
typedef __attribute__((ext_vector_type(8)))  float  v8f;

#define IN_F  256
#define OUT_F 128

__global__ void zero_f32_kernel(float* __restrict__ p, int n) {
    int i = blockIdx.x * blockDim.x + threadIdx.x;
    if (i < n) p[i] = 0.0f;
}

// z[n, o] = sum_k h[n, k] * W[o, k]   (h: [N,256] f32, W: [128,256] f32, z: [N,128] f32)
// bf16 WMMA 16x16x32, fp32 accumulate. One block = 16 rows x 128 cols, wave w -> col tile w.
__global__ __launch_bounds__(256) void gemm_proj_kernel(const float* __restrict__ h,
                                                        const float* __restrict__ W,
                                                        float* __restrict__ z) {
    const int wave = threadIdx.x >> 5;
    const int lane = threadIdx.x & 31;
    const int half = lane >> 4;         // 0: lanes 0-15, 1: lanes 16-31
    const int l16  = lane & 15;

    const int m0 = blockIdx.x * 16;     // row tile base
    const int o0 = wave * 16;           // col tile base

    const int m = m0 + l16;             // A-matrix row owned by this lane
    const int o = o0 + l16;             // B-matrix column owned by this lane
    const int akb = half * 16;          // A K-offset within 32-chunk (contiguous, matches B)
    const int bkb = half * 16;          // B K-offset within 32-chunk (contiguous, matches A)

    const float* __restrict__ hrow = h + (long)m * IN_F;
    const float* __restrict__ wrow = W + (long)o * IN_F;

    v8f acc = {};

    #pragma unroll
    for (int k0 = 0; k0 < IN_F; k0 += 32) {
        // ---- A tile: lane holds K = k0+akb+0..15 in elems 0..15
        const float4* ap0 = (const float4*)(hrow + k0 + akb);
        const float4* ap1 = (const float4*)(hrow + k0 + akb + 8);
        float4 a0 = ap0[0], a1 = ap0[1];
        float4 a2 = ap1[0], a3 = ap1[1];
        v16bf a;
        a[0]  = (__bf16)a0.x; a[1]  = (__bf16)a0.y; a[2]  = (__bf16)a0.z; a[3]  = (__bf16)a0.w;
        a[4]  = (__bf16)a1.x; a[5]  = (__bf16)a1.y; a[6]  = (__bf16)a1.z; a[7]  = (__bf16)a1.w;
        a[8]  = (__bf16)a2.x; a[9]  = (__bf16)a2.y; a[10] = (__bf16)a2.z; a[11] = (__bf16)a2.w;
        a[12] = (__bf16)a3.x; a[13] = (__bf16)a3.y; a[14] = (__bf16)a3.z; a[15] = (__bf16)a3.w;

        // ---- B tile: lane holds column o, K = k0+bkb+0..15 in elems 0..15
        const float4* bp = (const float4*)(wrow + k0 + bkb);
        float4 b0 = bp[0], b1 = bp[1], b2 = bp[2], b3 = bp[3];
        v16bf b;
        b[0]  = (__bf16)b0.x; b[1]  = (__bf16)b0.y; b[2]  = (__bf16)b0.z; b[3]  = (__bf16)b0.w;
        b[4]  = (__bf16)b1.x; b[5]  = (__bf16)b1.y; b[6]  = (__bf16)b1.z; b[7]  = (__bf16)b1.w;
        b[8]  = (__bf16)b2.x; b[9]  = (__bf16)b2.y; b[10] = (__bf16)b2.z; b[11] = (__bf16)b2.w;
        b[12] = (__bf16)b3.x; b[13] = (__bf16)b3.y; b[14] = (__bf16)b3.z; b[15] = (__bf16)b3.w;

        // (neg_a, A, neg_b, B, c_mod, C, reuse_a, reuse_b)
        acc = __builtin_amdgcn_wmma_f32_16x16x32_bf16(false, a, false, b,
                                                      (short)0, acc, false, false);
    }

    // ---- C/D store: VGPR r -> row = m0 + half*8 + r, col = o0 + l16
    const int rbase = m0 + half * 8;
    float* __restrict__ zcol = z + (long)rbase * OUT_F + o0 + l16;
    #pragma unroll
    for (int r = 0; r < 8; ++r) {
        zcol[(long)r * OUT_F] = acc[r];
    }
}

// One wave32 per edge: gather z[src] (float4/lane), scatter-add into out[dst] + degree count.
// z and out both fit in the 192MB L2, so this phase runs at L2 bandwidth.
__global__ __launch_bounds__(256) void edge_scatter_kernel(const float* __restrict__ z,
                                                           const int* __restrict__ src,
                                                           const int* __restrict__ dst,
                                                           float* __restrict__ out,
                                                           float* __restrict__ deg,
                                                           int n_edges) {
    long gid  = (long)blockIdx.x * blockDim.x + threadIdx.x;
    int  e    = (int)(gid >> 5);
    int  lane = (int)(gid & 31);
    if (e >= n_edges) return;

    int s = src[e];
    int d = dst[e];

    float4 v = ((const float4*)(z + (long)s * OUT_F))[lane];

    float* op = out + (long)d * OUT_F + lane * 4;
    __hip_atomic_fetch_add(op + 0, v.x, __ATOMIC_RELAXED, __HIP_MEMORY_SCOPE_AGENT);
    __hip_atomic_fetch_add(op + 1, v.y, __ATOMIC_RELAXED, __HIP_MEMORY_SCOPE_AGENT);
    __hip_atomic_fetch_add(op + 2, v.z, __ATOMIC_RELAXED, __HIP_MEMORY_SCOPE_AGENT);
    __hip_atomic_fetch_add(op + 3, v.w, __ATOMIC_RELAXED, __HIP_MEMORY_SCOPE_AGENT);
    if (lane == 0) {
        __hip_atomic_fetch_add(deg + d, 1.0f, __ATOMIC_RELAXED, __HIP_MEMORY_SCOPE_AGENT);
    }
}

__global__ void finalize_kernel(float* __restrict__ out, const float* __restrict__ deg, int total) {
    int i = blockIdx.x * blockDim.x + threadIdx.x;
    if (i < total) {
        int n = i >> 7;  // / OUT_F
        out[i] = out[i] / fmaxf(deg[n], 1.0f);
    }
}

extern "C" void kernel_launch(void* const* d_in, const int* in_sizes, int n_in,
                              void* d_out, int out_size, void* d_ws, size_t ws_size,
                              hipStream_t stream) {
    const float* h   = (const float*)d_in[0];   // [N, 256] f32
    const float* W   = (const float*)d_in[1];   // [128, 256] f32
    const int*   src = (const int*)d_in[2];     // [E]
    const int*   dst = (const int*)d_in[3];     // [E]
    float*       out = (float*)d_out;           // [N, 128] f32

    const int n_nodes = in_sizes[0] / IN_F;
    const int n_edges = in_sizes[2];

    // workspace: z [N,128] f32, then deg [N] f32
    float* z   = (float*)d_ws;
    float* deg = z + (size_t)n_nodes * OUT_F;

    const int total_out = n_nodes * OUT_F;

    // zero accumulators every call (deterministic; atomics accumulate below)
    zero_f32_kernel<<<(total_out + 255) / 256, 256, 0, stream>>>(out, total_out);
    zero_f32_kernel<<<(n_nodes + 255) / 256, 256, 0, stream>>>(deg, n_nodes);

    // projection GEMM: 6250 blocks, each 16 rows x 128 cols
    gemm_proj_kernel<<<n_nodes / 16, 256, 0, stream>>>(h, W, z);

    // edge phase: one wave per edge
    long ethreads = (long)n_edges * 32;
    edge_scatter_kernel<<<(int)((ethreads + 255) / 256), 256, 0, stream>>>(
        z, src, dst, out, deg, n_edges);

    // mean over mailbox
    finalize_kernel<<<(total_out + 255) / 256, 256, 0, stream>>>(out, deg, total_out);
}
